// Self_Attn_8967891714323
// MI455X (gfx1250) — matrix operand order repaired and checked by reading, not compiled, not run on hardware
//
#include <hip/hip_runtime.h>

typedef __attribute__((ext_vector_type(16))) _Float16 v16h;
typedef __attribute__((ext_vector_type(8)))  _Float16 v8h;
typedef __attribute__((ext_vector_type(8)))  float    v8f;

#define BB 4
#define CC 64
#define DD 8
#define NN 4096

union U16 { v16h v; _Float16 h[16]; unsigned u[8]; };
union U8  { v8h  v; _Float16 h[8];  unsigned u[4]; };

// ------------------------------------------------------------------
// Kernel 1: 1x1-conv projections q,k,v from x; emit f16 in WMMA-friendly
// layouts: qh/kh = [B][N][8] (row i -> 8 contiguous halves = one A/B lane
// K-pack), vt = [B][C][N] (row c -> contiguous j for PV A-tile lane loads).
// ------------------------------------------------------------------
__global__ __launch_bounds__(256) void proj_kernel(
    const float* __restrict__ x,
    const float* __restrict__ Wq, const float* __restrict__ bq,
    const float* __restrict__ Wk, const float* __restrict__ bk,
    const float* __restrict__ Wv, const float* __restrict__ bv,
    _Float16* __restrict__ qh, _Float16* __restrict__ kh,
    _Float16* __restrict__ vt)
{
    __shared__ float sWq[DD * CC];
    __shared__ float sWk[DD * CC];
    __shared__ float sWv[CC * CC];

    const int t = threadIdx.x;
    for (int i = t; i < DD * CC; i += 256) { sWq[i] = Wq[i]; sWk[i] = Wk[i]; }
    for (int i = t; i < CC * CC; i += 256) { sWv[i] = Wv[i]; }
    __syncthreads();

    const int gid = blockIdx.x * 256 + t;
    const int b = gid >> 12;          // N = 4096 pixels per batch
    const int n = gid & (NN - 1);

    float xv[CC];
    const float* xb = x + (size_t)(b * CC) * NN + n;
#pragma unroll
    for (int c = 0; c < CC; ++c) xv[c] = xb[(size_t)c * NN];

#pragma unroll
    for (int d = 0; d < DD; ++d) {
        float aq = bq[d], ak = bk[d];
#pragma unroll
        for (int c = 0; c < CC; ++c) {
            aq = fmaf(sWq[d * CC + c], xv[c], aq);
            ak = fmaf(sWk[d * CC + c], xv[c], ak);
        }
        qh[((size_t)(b * NN + n)) * DD + d] = (_Float16)aq;
        kh[((size_t)(b * NN + n)) * DD + d] = (_Float16)ak;
    }

    for (int e = 0; e < CC; ++e) {
        float av = bv[e];
#pragma unroll
        for (int c = 0; c < CC; ++c) av = fmaf(sWv[e * CC + c], xv[c], av);
        vt[((size_t)(b * CC + e)) * NN + n] = (_Float16)av;
    }
}

// ------------------------------------------------------------------
// Kernel 2: flash-attention, 32 keys per iteration.
//  - 2 QK WMMAs -> E^T for j 0..15 / 16..31 (lane = query col, 8 rows each)
//  - online softmax: lane-local over 16 + one shfl_xor(16) each for max/sum
//  - P repacked to full-K=32 B-layout (no zero halves)
//  - 4 PV WMMAs at K=32 fully utilized accumulate out^T[c,i]
// ------------------------------------------------------------------
__global__ __launch_bounds__(256) void attn_kernel(
    const _Float16* __restrict__ qh, const _Float16* __restrict__ kh,
    const _Float16* __restrict__ vt,
    const float* __restrict__ x, const float* __restrict__ gamma,
    float* __restrict__ out)
{
    const int lane  = threadIdx.x & 31;
    const int wid   = threadIdx.x >> 5;
    const int qtile = blockIdx.x * 8 + wid;
    const int b  = qtile >> 8;              // 256 query tiles per batch
    const int i0 = (qtile & 255) * 16;
    const int m  = lane & 15;
    const int g  = lane >> 4;

    // B operand: Q tile (K=d x N=i). K=0..7 live in lane-group 0 halves 0..7.
    U16 bqm;
#pragma unroll
    for (int r = 0; r < 16; ++r) bqm.h[r] = (_Float16)0.f;
    if (g == 0) {
        const v8h qrow = *(const v8h*)(qh + ((size_t)(b * NN) + i0 + m) * DD);
#pragma unroll
        for (int r = 0; r < 8; ++r) bqm.h[r] = qrow[r];
    }

    v8f acc[4];
#pragma unroll
    for (int tt = 0; tt < 4; ++tt)
#pragma unroll
        for (int r = 0; r < 8; ++r) acc[tt][r] = 0.0f;

    float m_run = -3.0e38f;
    float l_run = 0.0f;

    const _Float16* khb = kh + (size_t)(b * NN) * DD;
    const _Float16* vtb = vt + (size_t)(b * CC) * NN;

    for (int j0 = 0; j0 < NN; j0 += 32) {
        if (j0 + 32 < NN)
            __builtin_prefetch(khb + ((size_t)(j0 + 32 + m)) * DD, 0, 3);

        // A operands: two K tiles (rows j, K=d), zero-padded to K=32.
        U16 ak0, ak1;
#pragma unroll
        for (int r = 0; r < 16; ++r) { ak0.h[r] = (_Float16)0.f; ak1.h[r] = (_Float16)0.f; }
        if (g == 0) {
            const v8h k0 = *(const v8h*)(khb + (size_t)(j0 + m) * DD);
            const v8h k1 = *(const v8h*)(khb + (size_t)(j0 + 16 + m) * DD);
#pragma unroll
            for (int r = 0; r < 8; ++r) { ak0.h[r] = k0[r]; ak1.h[r] = k1[r]; }
        }

        v8f cz = {};
        v8f et0 = __builtin_amdgcn_wmma_f32_16x16x32_f16(
            false, ak0.v, false, bqm.v, (short)0, cz, false, false);
        v8f et1 = __builtin_amdgcn_wmma_f32_16x16x32_f16(
            false, ak1.v, false, bqm.v, (short)0, cz, false, false);
        // et0: rows j = j0 + r + 8*g;  et1: rows j = j0 + 16 + r + 8*g.

        float tmax = et0[0];
#pragma unroll
        for (int r = 1; r < 8; ++r) tmax = fmaxf(tmax, et0[r]);
#pragma unroll
        for (int r = 0; r < 8; ++r) tmax = fmaxf(tmax, et1[r]);
        tmax = fmaxf(tmax, __shfl_xor(tmax, 16, 32));
        const float m_new = fmaxf(m_run, tmax);
        const float scale = __expf(m_run - m_new);

        float p0[8], p1[8], tsum = 0.0f;
#pragma unroll
        for (int r = 0; r < 8; ++r) { p0[r] = __expf(et0[r] - m_new); tsum += p0[r]; }
#pragma unroll
        for (int r = 0; r < 8; ++r) { p1[r] = __expf(et1[r] - m_new); tsum += p1[r]; }
        tsum += __shfl_xor(tsum, 16, 32);
        l_run = l_run * scale + tsum;
        m_run = m_new;

#pragma unroll
        for (int tt = 0; tt < 4; ++tt)
#pragma unroll
            for (int r = 0; r < 8; ++r) acc[tt][r] *= scale;

        // Repack P^T -> full K=32 B-layout:
        //   lane group 0: halves 0..7 = own p_lo (j0+0..7), 8..15 = partner p_lo (j0+8..15)
        //   lane group 1: halves 0..7 = partner p_hi (j0+16..23), 8..15 = own p_hi (j0+24..31)
        U8 plo, phi;
#pragma unroll
        for (int r = 0; r < 8; ++r) { plo.h[r] = (_Float16)p0[r]; phi.h[r] = (_Float16)p1[r]; }
        unsigned qlo[4], qhi[4];
#pragma unroll
        for (int tt = 0; tt < 4; ++tt) {
            qlo[tt] = (unsigned)__shfl_xor((int)plo.u[tt], 16, 32);
            qhi[tt] = (unsigned)__shfl_xor((int)phi.u[tt], 16, 32);
        }
        U16 bp;
        if (g == 0) {
#pragma unroll
            for (int tt = 0; tt < 4; ++tt) { bp.u[tt] = plo.u[tt]; bp.u[tt + 4] = qlo[tt]; }
        } else {
#pragma unroll
            for (int tt = 0; tt < 4; ++tt) { bp.u[tt] = qhi[tt]; bp.u[tt + 4] = phi.u[tt]; }
        }

        // PV: out^T[c,i] += V[c,j] * P^T[j,i]; A rows c, full K=32 of j:
        //   halves 0..7  = j0 + 16*g + 0..7   (K 0..7 / 16..23)
        //   halves 8..15 = j0 + 16*g + 8..15  (K 8..15 / 24..31)
        const _Float16* vbase = vtb + j0 + 16 * g;
#pragma unroll
        for (int tt = 0; tt < 4; ++tt) {
            const _Float16* vrowp = vbase + (size_t)(tt * 16 + m) * NN;
            const v8h vlo = *(const v8h*)(vrowp);
            const v8h vhi = *(const v8h*)(vrowp + 8);
            U16 av;
#pragma unroll
            for (int r = 0; r < 8; ++r) { av.h[r] = vlo[r]; av.h[r + 8] = vhi[r]; }
            acc[tt] = __builtin_amdgcn_wmma_f32_16x16x32_f16(
                false, av.v, false, bp.v, (short)0, acc[tt], false, false);
        }
    }

    // Epilogue: normalize, gamma*out + x.
    const float inv_l = 1.0f / l_run;
    const float gma = gamma[0];
    const int i = i0 + m;
#pragma unroll
    for (int tt = 0; tt < 4; ++tt) {
#pragma unroll
        for (int r = 0; r < 8; ++r) {
            const int c = tt * 16 + r + g * 8;
            const size_t idx = ((size_t)(b * CC + c)) * NN + i;
            out[idx] = gma * acc[tt][r] * inv_l + x[idx];
        }
    }
}

extern "C" void kernel_launch(void* const* d_in, const int* in_sizes, int n_in,
                              void* d_out, int out_size, void* d_ws, size_t ws_size,
                              hipStream_t stream)
{
    const float* x     = (const float*)d_in[0];
    const float* Wq    = (const float*)d_in[1];
    const float* bq    = (const float*)d_in[2];
    const float* Wk    = (const float*)d_in[3];
    const float* bk    = (const float*)d_in[4];
    const float* Wv    = (const float*)d_in[5];
    const float* bv    = (const float*)d_in[6];
    const float* gamma = (const float*)d_in[7];
    float* out = (float*)d_out;

    _Float16* qh = (_Float16*)d_ws;                    // [B][N][8]  f16
    _Float16* kh = qh + (size_t)BB * NN * DD;          // [B][N][8]  f16
    _Float16* vt = kh + (size_t)BB * NN * DD;          // [B][C][N]  f16
    // total workspace: 2 * 256KB + 2MB = 2.5 MB

    proj_kernel<<<(BB * NN) / 256, 256, 0, stream>>>(x, Wq, bq, Wk, bk, Wv, bv,
                                                     qh, kh, vt);
    attn_kernel<<<(BB * 256) / 8, 256, 0, stream>>>(qh, kh, vt, x, gamma, out);
}
